// Qwen3Attention_11278584119928
// MI455X (gfx1250) — matrix operand order repaired and checked by reading, not compiled, not run on hardware
//
#include <hip/hip_runtime.h>
#include <hip/hip_bf16.h>

// ---------------------------------------------------------------------------
// Qwen3-style attention, CDNA5 (gfx1250, wave32, WMMA bf16).
//   B=1, S=4096, HID=2048, NH=16, NKV=8, HD=128, GROUPS=2, theta=1e6, eps=1e-6
// All GEMMs via v_wmma_f32_16x16x32_bf16 (fp32 accumulate).
// Wave-level N-blocking: each wave computes a 16x64 strip (GEMMs) or two
// 16x16 M-tiles (attention) so A/B fragments are reused across WMMAs.
// ---------------------------------------------------------------------------

typedef __bf16 bf16_t;
typedef __attribute__((ext_vector_type(16))) __bf16 v16bf;
typedef __attribute__((ext_vector_type(8)))  float  v8f;

#define S_LEN 4096
#define HIDDEN 2048
#define N_HEADS 16
#define N_KV 8
#define HEAD_D 128
#define INV_SQRT_D 0.08838834764831845f
#define RMS_EPS 1e-6f
#define ROPE_THETA 1000000.0f

#define WMMA_BF16(a, b, c) \
  __builtin_amdgcn_wmma_f32_16x16x32_bf16(false, (a), false, (b), (short)0, (c), false, false)

// ---- fragment loaders ------------------------------------------------------
// A-matrix 16x32 bf16 (ISA 7.12.2): lane m=l&15; lanes 0-15 hold K {0..7,16..23},
// lanes 16-31 hold K {8..15,24..31}; -> two 16B contiguous loads per lane.
static __device__ __forceinline__ v16bf load_fragA(const bf16_t* base, int stride, int lane) {
  int m   = lane & 15;
  int sel = (lane >> 4) << 3;                 // 0 or 8
  const bf16_t* p = base + (size_t)m * stride + sel;
  union { v16bf v; uint4 u[2]; } f;
  f.u[0] = *(const uint4*)(p);
  f.u[1] = *(const uint4*)(p + 16);
  return f.v;
}
// B-matrix 32x16 bf16: lane n=l&15; lanes 0-15 hold K {0..7,16..23}, lanes 16-31
// hold K {8..15,24..31} -> two 16B contiguous loads per lane (matches A).
static __device__ __forceinline__ v16bf load_fragB(const bf16_t* base, int stride, int lane) {
  int n  = lane & 15;
  int kg = (lane >> 4) << 3;                  // 0 or 8
  const bf16_t* p = base + (size_t)n * stride + kg;
  union { v16bf v; uint4 u[2]; } f;
  f.u[0] = *(const uint4*)(p);
  f.u[1] = *(const uint4*)(p + 16);
  return f.v;
}

// ---- fp32 -> bf16 convert (vector x4) --------------------------------------
__global__ void f2bf4_kernel(const float4* __restrict__ src, uint2* __restrict__ dst, int n4) {
  int i = blockIdx.x * blockDim.x + threadIdx.x;
  if (i < n4) {
    float4 v = src[i];
    union { bf16_t b[4]; uint2 u; } o;
    o.b[0] = (bf16_t)v.x; o.b[1] = (bf16_t)v.y;
    o.b[2] = (bf16_t)v.z; o.b[3] = (bf16_t)v.w;
    dst[i] = o.u;
  }
}

// ---- fused QKV projection + RMSNorm + RoPE ---------------------------------
// grid: (S/16) row-tiles x 8 col-blocks of 512. block: 256 = 8 waves.
// Wave w computes a 16x64 strip (4 WMMA tiles, one shared A fragment per
// K-step). Col-blocks 0-3 -> Q heads, 4-5 -> K heads, 6-7 -> V heads.
// Q/K: RMSNorm(dim=128) + RoPE, stored bf16 [h][s][d].
// V: stored TRANSPOSED bf16 [hkv][d][s] so PV B-fragments are contiguous.
__global__ __launch_bounds__(256) void qkv_kernel(
    const bf16_t* __restrict__ hs, const bf16_t* __restrict__ wq,
    const bf16_t* __restrict__ wk, const bf16_t* __restrict__ wv,
    const float* __restrict__ qnw, const float* __restrict__ knw,
    bf16_t* __restrict__ qout, bf16_t* __restrict__ kout, bf16_t* __restrict__ vtout) {
  int cb = blockIdx.x & 7;
  int s0 = (blockIdx.x >> 3) << 4;
  int wave = threadIdx.x >> 5, lane = threadIdx.x & 31;

  int colbase = cb * 512 + wave * 64;          // global col of wave's strip
  const bf16_t* Wrow;
  if (colbase < 2048)      Wrow = wq + (size_t)colbase * HIDDEN;
  else if (colbase < 3072) Wrow = wk + (size_t)(colbase - 2048) * HIDDEN;
  else                     Wrow = wv + (size_t)(colbase - 3072) * HIDDEN;
  const bf16_t* A0 = hs + (size_t)s0 * HIDDEN;

  v8f acc[4] = {};
  for (int k = 0; k < HIDDEN; k += 32) {
    __builtin_prefetch(A0 + k + 128, 0, 1);
    __builtin_prefetch(Wrow + k + 128, 0, 1);
    v16bf a = load_fragA(A0 + k, HIDDEN, lane);          // shared across 4 tiles
#pragma unroll
    for (int t = 0; t < 4; t++) {
      v16bf b = load_fragB(Wrow + (size_t)(t * 16) * HIDDEN + k, HIDDEN, lane);
      acc[t] = WMMA_BF16(a, b, acc[t]);
    }
  }

  __shared__ float tile_f[16][520];                      // 16 x 512 (+pad)
#pragma unroll
  for (int t = 0; t < 4; t++)
#pragma unroll
    for (int i = 0; i < 8; i++) {
      int m = i + ((lane >> 4) << 3);
      tile_f[m][wave * 64 + t * 16 + (lane & 15)] = acc[t][i];
    }
  __syncthreads();

  int kind = (cb < 4) ? 0 : (cb < 6) ? 1 : 2;

  if (kind == 2) {                       // V: bf16 transpose store [d][s]
#pragma unroll
    for (int rep = 0; rep < 2; rep++) {
      int dg = threadIdx.x + rep * 256;                  // 0..511 within block
      int hl = dg >> 7, d = dg & 127;
      int hv = (cb - 6) * 4 + hl;
      bf16_t* dst = vtout + ((size_t)hv * HEAD_D + d) * S_LEN + s0;
#pragma unroll
      for (int j = 0; j < 16; j++) dst[j] = (bf16_t)tile_f[j][dg];
    }
    return;
  }

  // Q/K: per-(head,row) RMSNorm over 128 then RoPE. 4 threads per (head,row),
  // 32 cols each: c = cl + 4j; RoPE pair (c, c+64) is (j, j+16), SAME thread.
  int g = threadIdx.x >> 2;                              // 0..63 = head*16+row
  int cl = threadIdx.x & 3;
  int hl = g >> 4, row = g & 15;
  int pos = s0 + row;
  float x[32], ss = 0.f;
#pragma unroll
  for (int j = 0; j < 32; j++) {
    x[j] = tile_f[row][hl * 128 + cl + 4 * j];
    ss += x[j] * x[j];
  }
#pragma unroll
  for (int o = 1; o < 4; o <<= 1) ss += __shfl_xor(ss, o, 4);
  float scale = rsqrtf(ss * (1.0f / 128.0f) + RMS_EPS);
  const float* nw = (kind == 0) ? qnw : knw;
#pragma unroll
  for (int j = 0; j < 32; j++) x[j] = x[j] * scale * nw[cl + 4 * j];

  float y[32];
#pragma unroll
  for (int j = 0; j < 16; j++) {
    int c = cl + 4 * j;                                  // c in [0,64)
    float fr  = __powf(ROPE_THETA, -(float)c * (1.0f / 64.0f));
    float ang = (float)pos * fr;
    float co = cosf(ang), si = sinf(ang);
    y[j]      = x[j] * co - x[j + 16] * si;              // q*cos + rot_half(q)*sin
    y[j + 16] = x[j + 16] * co + x[j] * si;
  }
  int hd = (kind == 0) ? (cb * 4 + hl) : ((cb - 4) * 4 + hl);
  bf16_t* dst = ((kind == 0) ? qout : kout) + ((size_t)hd * S_LEN + pos) * HEAD_D;
#pragma unroll
  for (int j = 0; j < 32; j++) dst[cl + 4 * j] = (bf16_t)y[j];
}

// ---- flash attention (causal, GQA), 32-query tiles -------------------------
// grid: NH x (S/32). block: 256 = 8 waves. Wave w owns key-slice / d-slice
// [16w,16w+16) and computes TWO 16x16 M-tiles, so every K/V B-fragment load
// feeds two WMMAs. Streams 128-key blocks with online softmax in LDS.
__global__ __launch_bounds__(256) void attn_kernel(
    const bf16_t* __restrict__ qb, const bf16_t* __restrict__ kb,
    const bf16_t* __restrict__ vtb, bf16_t* __restrict__ attn_out) {
  int h  = blockIdx.x & (N_HEADS - 1);
  int s0 = (blockIdx.x >> 4) << 5;                       // 32-row query tile
  int hkv = h >> 1;                                      // repeat_interleave(2)
  int wave = threadIdx.x >> 5, lane = threadIdx.x & 31;
  int srow = threadIdx.x >> 3, scol = threadIdx.x & 7;   // softmax: 8 thr/row

  const bf16_t* Q  = qb  + ((size_t)h   * S_LEN + s0) * HEAD_D;
  const bf16_t* K  = kb  + (size_t)hkv * S_LEN * HEAD_D;
  const bf16_t* Vt = vtb + (size_t)hkv * HEAD_D * S_LEN;

  __shared__ float sc[32][130];
  __shared__ __align__(16) bf16_t pr[32][136];
  __shared__ float alpha_l[32];
  __shared__ float lsum_l[32];

  v16bf qa0[4], qa1[4];
#pragma unroll
  for (int s = 0; s < 4; s++) {
    qa0[s] = load_fragA(Q + 32 * s, HEAD_D, lane);
    qa1[s] = load_fragA(Q + (size_t)16 * HEAD_D + 32 * s, HEAD_D, lane);
  }

  v8f acc0 = {}, acc1 = {};
  float mrun = -1e30f, lrun = 0.f;
  int nkb = (s0 + 32 + 127) >> 7;                        // causal key-block count

  for (int blk = 0; blk < nkb; blk++) {
    int kbase = blk << 7;

    // --- scores: wave w -> keys [kbase+16w,+16), both 16-row M-tiles --------
    v8f sa0 = {}, sa1 = {};
    const bf16_t* Kp = K + (size_t)(kbase + wave * 16) * HEAD_D;
#pragma unroll
    for (int s = 0; s < 4; s++) {
      v16bf b = load_fragB(Kp + 32 * s, HEAD_D, lane);   // B[k=d][n=keypos]
      sa0 = WMMA_BF16(qa0[s], b, sa0);
      sa1 = WMMA_BF16(qa1[s], b, sa1);
    }
#pragma unroll
    for (int i = 0; i < 8; i++) {
      int m = i + ((lane >> 4) << 3);
      sc[m][wave * 16 + (lane & 15)]      = sa0[i];
      sc[m + 16][wave * 16 + (lane & 15)] = sa1[i];
    }
    __syncthreads();

    // --- online softmax over this 32x128 block ------------------------------
    float pv[16], mx = -1e30f;
#pragma unroll
    for (int j = 0; j < 16; j++) {
      int c = scol + 8 * j;
      float v = sc[srow][c] * INV_SQRT_D;
      if (kbase + c > s0 + srow) v = -1e30f;             // causal mask
      pv[j] = v; mx = fmaxf(mx, v);
    }
#pragma unroll
    for (int o = 1; o < 8; o <<= 1) mx = fmaxf(mx, __shfl_xor(mx, o, 8));
    float mnew = fmaxf(mrun, mx);
    float al   = __expf(mrun - mnew);
    float ps   = 0.f;
#pragma unroll
    for (int j = 0; j < 16; j++) {
      float e = __expf(pv[j] - mnew);
      ps += e;
      pr[srow][scol + 8 * j] = (bf16_t)e;
    }
#pragma unroll
    for (int o = 1; o < 8; o <<= 1) ps += __shfl_xor(ps, o, 8);
    lrun = al * lrun + ps;
    mrun = mnew;
    if (scol == 0) alpha_l[srow] = al;
    __syncthreads();

    // --- PV: wave w -> d-slice [16w,+16), both M-tiles ----------------------
#pragma unroll
    for (int i = 0; i < 8; i++) {
      int m = i + ((lane >> 4) << 3);
      acc0[i] *= alpha_l[m];
      acc1[i] *= alpha_l[m + 16];
    }
    const bf16_t* Vp = Vt + (size_t)(wave * 16) * S_LEN + kbase;
#pragma unroll
    for (int s = 0; s < 4; s++) {
      v16bf a0 = load_fragA(&pr[0][0] + 32 * s, 136, lane);   // probs rows 0-15
      v16bf a1 = load_fragA(&pr[16][0] + 32 * s, 136, lane);  // probs rows 16-31
      v16bf b  = load_fragB(Vp + 32 * s, S_LEN, lane);        // B[k=pos][n=d]
      acc0 = WMMA_BF16(a0, b, acc0);
      acc1 = WMMA_BF16(a1, b, acc1);
    }
    __syncthreads();
  }

  if (scol == 0) lsum_l[srow] = lrun;
  __syncthreads();

#pragma unroll
  for (int i = 0; i < 8; i++) {
    int m = i + ((lane >> 4) << 3);
    float o0 = acc0[i] * (1.0f / lsum_l[m]);
    float o1 = acc1[i] * (1.0f / lsum_l[m + 16]);
    size_t colo = (size_t)h * HEAD_D + wave * 16 + (lane & 15);
    attn_out[(size_t)(s0 + m) * HIDDEN + colo]      = (bf16_t)o0;
    attn_out[(size_t)(s0 + m + 16) * HIDDEN + colo] = (bf16_t)o1;
  }
}

// ---- output projection: out[s][e] = sum_f attn[s][f] * wo[e][f] ------------
// grid: (S/16) x 4 col-blocks of 512; wave computes 16x64 strip.
__global__ __launch_bounds__(256) void oproj_kernel(
    const bf16_t* __restrict__ attn, const bf16_t* __restrict__ wo,
    float* __restrict__ out) {
  int cb = blockIdx.x & 3;
  int s0 = (blockIdx.x >> 2) << 4;
  int wave = threadIdx.x >> 5, lane = threadIdx.x & 31;
  int colbase = cb * 512 + wave * 64;
  const bf16_t* A0 = attn + (size_t)s0 * HIDDEN;
  const bf16_t* Wn = wo + (size_t)colbase * HIDDEN;
  v8f acc[4] = {};
  for (int k = 0; k < HIDDEN; k += 32) {
    __builtin_prefetch(A0 + k + 128, 0, 1);
    __builtin_prefetch(Wn + k + 128, 0, 1);
    v16bf a = load_fragA(A0 + k, HIDDEN, lane);
#pragma unroll
    for (int t = 0; t < 4; t++) {
      v16bf b = load_fragB(Wn + (size_t)(t * 16) * HIDDEN + k, HIDDEN, lane);
      acc[t] = WMMA_BF16(a, b, acc[t]);
    }
  }
#pragma unroll
  for (int t = 0; t < 4; t++)
#pragma unroll
    for (int i = 0; i < 8; i++) {
      int m = i + ((lane >> 4) << 3);
      out[(size_t)(s0 + m) * HIDDEN + colbase + t * 16 + (lane & 15)] = acc[t][i];
    }
}

// ---------------------------------------------------------------------------
extern "C" void kernel_launch(void* const* d_in, const int* in_sizes, int n_in,
                              void* d_out, int out_size, void* d_ws, size_t ws_size,
                              hipStream_t stream) {
  const float* hs  = (const float*)d_in[0];
  // d_in[1] = attention_mask: pure causal -> synthesized analytically, never read.
  const float* wq  = (const float*)d_in[2];
  const float* wk  = (const float*)d_in[3];
  const float* wv  = (const float*)d_in[4];
  const float* wo  = (const float*)d_in[5];
  const float* qnw = (const float*)d_in[6];
  const float* knw = (const float*)d_in[7];
  float* out = (float*)d_out;

  char* ws = (char*)d_ws;
  size_t off = 0;
  auto carve = [&](size_t bytes) { void* p = ws + off; off += (bytes + 255) & ~(size_t)255; return p; };

  bf16_t* hs_b   = (bf16_t*)carve((size_t)S_LEN * HIDDEN * 2);
  bf16_t* wq_b   = (bf16_t*)carve((size_t)N_HEADS * HEAD_D * HIDDEN * 2);
  bf16_t* wk_b   = (bf16_t*)carve((size_t)N_KV * HEAD_D * HIDDEN * 2);
  bf16_t* wv_b   = (bf16_t*)carve((size_t)N_KV * HEAD_D * HIDDEN * 2);
  bf16_t* wo_b   = (bf16_t*)carve((size_t)HIDDEN * N_HEADS * HEAD_D * 2);
  bf16_t* q_b    = (bf16_t*)carve((size_t)N_HEADS * S_LEN * HEAD_D * 2);
  bf16_t* k_b    = (bf16_t*)carve((size_t)N_KV * S_LEN * HEAD_D * 2);
  bf16_t* vt_b   = (bf16_t*)carve((size_t)N_KV * HEAD_D * S_LEN * 2);
  bf16_t* attn_b = (bf16_t*)carve((size_t)S_LEN * HIDDEN * 2);

  auto conv = [&](const float* s, bf16_t* d, int n) {
    int n4 = n >> 2;
    f2bf4_kernel<<<(n4 + 255) / 256, 256, 0, stream>>>((const float4*)s, (uint2*)d, n4);
  };
  conv(hs, hs_b, S_LEN * HIDDEN);
  conv(wq, wq_b, N_HEADS * HEAD_D * HIDDEN);
  conv(wk, wk_b, N_KV * HEAD_D * HIDDEN);
  conv(wv, wv_b, N_KV * HEAD_D * HIDDEN);
  conv(wo, wo_b, HIDDEN * N_HEADS * HEAD_D);

  // QKV + norm + rope: (S/16) row-tiles x 8 col-blocks of 512
  qkv_kernel<<<(S_LEN / 16) * 8, 256, 0, stream>>>(hs_b, wq_b, wk_b, wv_b, qnw, knw,
                                                   q_b, k_b, vt_b);
  // flash attention: NH x (S/32)
  attn_kernel<<<N_HEADS * (S_LEN / 32), 256, 0, stream>>>(q_b, k_b, vt_b, attn_b);
  // output projection: (S/16) x 4 col-blocks of 512
  oproj_kernel<<<(S_LEN / 16) * 4, 256, 0, stream>>>(attn_b, wo_b, out);
}
